// CausalSelfAttention_1056561955085
// MI455X (gfx1250) — matrix operand order repaired and checked by reading, not compiled, not run on hardware
//
#include <hip/hip_runtime.h>
#include <hip/hip_fp16.h>

typedef __attribute__((ext_vector_type(16))) _Float16 v16h;
typedef __attribute__((ext_vector_type(8)))  float    v8f;
typedef unsigned int u32;
typedef unsigned long long u64;
typedef __attribute__((ext_vector_type(4))) u32 u32x4;
typedef __attribute__((ext_vector_type(8))) u32 u32x8;

#define EMBED 1024
#define SEQN  2048
#define BATCH 2
#define HEADS 16
#define HD    64

union FragA { v16h h; u32 u[8]; };

// 16-byte async copy global -> LDS (per-lane addresses, tracked by ASYNCcnt)
static __device__ inline void async_cp16(u32 lds_byte, const void* gptr) {
    asm volatile("global_load_async_to_lds_b128 %0, %1, off"
                 :: "v"(lds_byte), "v"((u64)(uintptr_t)gptr) : "memory");
}
static __device__ inline u32 lds_off32(const void* p) {
    return (u32)(uintptr_t)p;   // generic LDS ptr: addr[31:0] is the LDS offset
}

// ---------------------------------------------------------------------------
// Elementwise f32 -> f16 convert
// ---------------------------------------------------------------------------
__global__ __launch_bounds__(256) void cvt_f16(const float* __restrict__ in,
                                               _Float16* __restrict__ out, int n) {
    int i = blockIdx.x * 256 + threadIdx.x;
    int stride = gridDim.x * 256;
    for (; i < n; i += stride) out[i] = (_Float16)in[i];
}

// ---------------------------------------------------------------------------
// Transpose + convert: out[n*K + k] = (f16) in[k*N + n]   (K = N = 1024)
// ---------------------------------------------------------------------------
__global__ __launch_bounds__(256) void transpose_cvt(const float* __restrict__ in,
                                                     _Float16* __restrict__ out) {
    __shared__ float tile[32][33];
    const int bn = blockIdx.x * 32;
    const int bk = blockIdx.y * 32;
    const int tx = threadIdx.x & 31;
    const int ty = threadIdx.x >> 5;
    for (int r = ty; r < 32; r += 8)
        tile[r][tx] = in[(size_t)(bk + r) * EMBED + bn + tx];
    __syncthreads();
    for (int r = ty; r < 32; r += 8)
        out[(size_t)(bn + r) * EMBED + bk + tx] = (_Float16)tile[tx][r];
}

// ---------------------------------------------------------------------------
// WMMA GEMM: C[M,N] = A[M,K] * Bt^T (+bias), A f16 [M,K], Bt f16 [N,K].
// Block 128x128, BK=32, 256 threads (8 waves), wave tile 32x64.
// Double-buffered async staging: issue next tile, s_wait_asynccnt 0x4
// (in-order completion per wave => previous tile landed), compute, repeat.
// ---------------------------------------------------------------------------
template<typename OutT, bool HASBIAS>
__global__ __launch_bounds__(256) void gemm_wmma_f16(
    const _Float16* __restrict__ A, const _Float16* __restrict__ Bt,
    const float* __restrict__ bias, OutT* __restrict__ C,
    int M, int N, int K)
{
    constexpr int BM = 128, BN = 128, BK = 32, BKP = 40;
    __shared__ _Float16 As[2][BM * BKP];   // [m][k]
    __shared__ _Float16 Bs[2][BN * BKP];   // [n][k]

    const int tid   = threadIdx.x;
    const int lane  = tid & 31;
    const int wid   = tid >> 5;
    const int wm    = wid >> 1;
    const int wn    = wid & 1;
    const int bm    = blockIdx.x * BM;
    const int bn    = blockIdx.y * BN;
    const int mhalf = lane >> 4;
    const int nlo   = lane & 15;
    const u32 asb   = lds_off32(As);
    const u32 bsb   = lds_off32(Bs);

    // per-thread staging chunk coords (2 chunks of 16B each per tile half)
    const int c0 = wid * 64 + lane, c1 = c0 + 32;
    const int r0 = c0 >> 2, p0 = c0 & 3;
    const int r1 = c1 >> 2, p1 = c1 & 3;
    const _Float16* Ab = A  + (size_t)bm * K;
    const _Float16* Bb = Bt + (size_t)bn * K;

    v8f acc[2][4];
    for (int i = 0; i < 2; ++i)
        for (int j = 0; j < 4; ++j)
            acc[i][j] = (v8f){0.f,0.f,0.f,0.f,0.f,0.f,0.f,0.f};

    auto stage = [&](int buf, int kb) {
        const u32 ab = asb + (u32)buf * BM * BKP * 2;
        const u32 bb = bsb + (u32)buf * BN * BKP * 2;
        async_cp16(ab + (u32)(r0 * BKP + p0 * 8) * 2, Ab + (size_t)r0 * K + kb + p0 * 8);
        async_cp16(ab + (u32)(r1 * BKP + p1 * 8) * 2, Ab + (size_t)r1 * K + kb + p1 * 8);
        async_cp16(bb + (u32)(r0 * BKP + p0 * 8) * 2, Bb + (size_t)r0 * K + kb + p0 * 8);
        async_cp16(bb + (u32)(r1 * BKP + p1 * 8) * 2, Bb + (size_t)r1 * K + kb + p1 * 8);
    };

    stage(0, 0);
    for (int kb = 0; kb < K; kb += BK) {
        const int buf = (kb >> 5) & 1;
        if (kb + BK < K) {
            stage(buf ^ 1, kb + BK);
            asm volatile("s_wait_asynccnt 0x4" ::: "memory");   // prev tile done
        } else {
            asm volatile("s_wait_asynccnt 0x0" ::: "memory");
        }
        __syncthreads();

        const _Float16* Ac = &As[buf][0];
        const _Float16* Bc = &Bs[buf][0];
        FragA afr[2], bfr[4];
        for (int i = 0; i < 2; ++i) {
            int mr = wm * 32 + i * 16 + nlo;
            int kbase = mhalf * 16;
            for (int v = 0; v < 8; ++v) {
                int k0 = (v << 1) + kbase;
                afr[i].u[v] = *(const u32*)(&Ac[mr * BKP + k0]);
            }
        }
        for (int j = 0; j < 4; ++j) {
            int nc = wn * 64 + j * 16 + nlo;
            int kb2 = mhalf * 16;
            for (int v = 0; v < 8; ++v)
                bfr[j].u[v] = *(const u32*)(&Bc[nc * BKP + kb2 + (v << 1)]);
        }
        for (int i = 0; i < 2; ++i)
            for (int j = 0; j < 4; ++j)
                acc[i][j] = __builtin_amdgcn_wmma_f32_16x16x32_f16(
                    false, afr[i].h, false, bfr[j].h,
                    (short)0, acc[i][j], false, false);
        __syncthreads();
    }

    for (int i = 0; i < 2; ++i)
        for (int j = 0; j < 4; ++j) {
            int m0 = bm + wm * 32 + i * 16 + mhalf * 8;
            int n  = bn + wn * 64 + j * 16 + nlo;
            float bz = 0.f;
            if constexpr (HASBIAS) bz = bias[n];
            for (int v = 0; v < 8; ++v)
                C[(size_t)(m0 + v) * N + n] = (OutT)(acc[i][j][v] + bz);
        }
}

// ---------------------------------------------------------------------------
// Flash attention, causal. Grid (SEQN/64, BATCH*HEADS), 128 threads (4 waves).
// K tiles double-buffered via Tensor Data Mover (wave 0 issues, scalar branch,
// s_wait_tensorcnt threshold 1 => previous tile landed). V staged manually
// (transposed), double-buffered.
// ---------------------------------------------------------------------------
__global__ __launch_bounds__(128) void flash_attn_wmma(
    const _Float16* __restrict__ q, const _Float16* __restrict__ k,
    const _Float16* __restrict__ v, _Float16* __restrict__ z)
{
    constexpr int KP = 72;  // lds_k [t][d] stride (TDM pad: 128B + 16B)
    constexpr int VP = 40;  // lds_v [d][t] stride
    constexpr int PP = 40;  // lds_p [s][t] stride
    __shared__ _Float16 lk[2][32 * KP];
    __shared__ _Float16 lv[2][64 * VP];
    __shared__ _Float16 lp[4][16 * PP];

    const int tid   = threadIdx.x;
    const int lane  = tid & 31;
    const int w     = tid >> 5;
    const int wS    = __builtin_amdgcn_readfirstlane(w);   // scalar wave id
    const int q0    = blockIdx.x * 64;
    const int bh    = blockIdx.y;
    const int b     = bh >> 4;
    const int h     = bh & 15;
    const size_t base = ((size_t)b * SEQN) * EMBED + (size_t)h * HD;
    const int qw    = q0 + w * 16;
    const int mhalf = lane >> 4;
    const int nlo   = lane & 15;
    const float scale = 1.0f / 32.0f;      // 1/sqrt(EMBED)

    // TDM D# group1 (constant): data_size=2B, pad 32DW interval / 4DW amount,
    // tensor_dim0=64, tensor_dim1=2048, tile 64x32, dim0 stride=1024.
    u32x8 g1;
    g1[0] = (1u << 16) | (1u << 20) | (4u << 22) | (3u << 25);
    g1[1] = (64u << 16);
    g1[2] = (2048u << 16);
    g1[3] = (64u << 16);
    g1[4] = 32u;
    g1[5] = 1024u;
    g1[6] = 0u;
    g1[7] = 0u;
    const u32 lk_lds0 = lds_off32(&lk[0][0]);
    const u32 lk_lds1 = lds_off32(&lk[1][0]);

    auto tdm_issue = [&](int t0, u32 lds_addr) {
        u64 ga = (u64)(uintptr_t)(k + base + (size_t)t0 * EMBED);
        u32x4 g0;
        g0[0] = 1u;                                            // count=1
        g0[1] = lds_addr;                                      // lds_addr
        g0[2] = (u32)ga;                                       // addr[31:0]
        g0[3] = (u32)((ga >> 32) & 0x01FFFFFFu) | (2u << 30);  // addr hi | type=2
        asm volatile("tensor_load_to_lds %0, %1" :: "s"(g0), "s"(g1) : "memory");
    };

    // ---- Q fragments (A-layout)
    FragA qf[2];
    {
        int mr = qw + nlo;
        int kbase = mhalf * 16;
        for (int f = 0; f < 2; ++f)
            for (int vv = 0; vv < 8; ++vv) {
                int d0 = f * 32 + (vv << 1) + kbase;
                qf[f].u[vv] = *(const u32*)(q + base + (size_t)mr * EMBED + d0);
            }
    }

    float mi[8], li[8];
    v8f accO[4];
    for (int vv = 0; vv < 8; ++vv) { mi[vv] = -__builtin_inff(); li[vv] = 0.f; }
    for (int jd = 0; jd < 4; ++jd) accO[jd] = (v8f){0.f,0.f,0.f,0.f,0.f,0.f,0.f,0.f};

    const int nTiles = (q0 + 64) / 32;

    if (wS == 0) tdm_issue(0, lk_lds0);            // prologue K tile 0

    for (int tb = 0; tb < nTiles; ++tb) {
        const int t0  = tb * 32;
        const int cur = tb & 1;
        // ---- V tile staged transposed [d][t] into lv[cur]
        for (int it = 0; it < 8; ++it) {
            int idx = tid + it * 128;
            int tr  = idx >> 5;
            int du  = idx & 31;
            u32 uv = *(const u32*)(v + base + (size_t)(t0 + tr) * EMBED + du * 2);
            union { u32 u; _Float16 hh[2]; } sp; sp.u = uv;
            lv[cur][(du * 2 + 0) * VP + tr] = sp.hh[0];
            lv[cur][(du * 2 + 1) * VP + tr] = sp.hh[1];
        }
        // ---- K pipeline: issue next tile, wait threshold for current
        if (wS == 0) {
            if (tb + 1 < nTiles) {
                tdm_issue(t0 + 32, (cur ^ 1) ? lk_lds1 : lk_lds0);
                __builtin_amdgcn_s_wait_tensorcnt(1);
            } else {
                __builtin_amdgcn_s_wait_tensorcnt(0);
            }
        }
        __syncthreads();

        const int activeS = __builtin_amdgcn_readfirstlane(t0 <= qw + 15 ? 1 : 0);
        if (activeS) {
            const _Float16* lkc = &lk[cur][0];
            const _Float16* lvc = &lv[cur][0];
            v8f sc[2];
            sc[0] = (v8f){0.f,0.f,0.f,0.f,0.f,0.f,0.f,0.f};
            sc[1] = (v8f){0.f,0.f,0.f,0.f,0.f,0.f,0.f,0.f};
            for (int ks = 0; ks < 2; ++ks)
                for (int j = 0; j < 2; ++j) {
                    FragA bf;
                    int tc  = j * 16 + nlo;
                    int kb2 = mhalf * 16;
                    for (int vv = 0; vv < 8; ++vv)
                        bf.u[vv] = *(const u32*)(&lkc[tc * KP + ks * 32 + kb2 + (vv << 1)]);
                    sc[j] = __builtin_amdgcn_wmma_f32_16x16x32_f16(
                        false, qf[ks].h, false, bf.h, (short)0, sc[j], false, false);
                }
            for (int j = 0; j < 2; ++j)
                for (int vv = 0; vv < 8; ++vv) {
                    int mrow = qw + mhalf * 8 + vv;
                    int tcol = t0 + j * 16 + nlo;
                    float s = sc[j][vv] * scale;
                    if (tcol > mrow) s = -1.0e30f;
                    sc[j][vv] = s;
                }
            float corr[8];
            for (int vv = 0; vv < 8; ++vv) {
                float mx = fmaxf(sc[0][vv], sc[1][vv]);
                for (int off = 8; off > 0; off >>= 1)
                    mx = fmaxf(mx, __shfl_xor(mx, off, 32));
                float mnew = fmaxf(mi[vv], mx);
                corr[vv] = __expf(mi[vv] - mnew);
                mi[vv] = mnew;
                float rs = 0.f;
                for (int j = 0; j < 2; ++j) {
                    float p = __expf(sc[j][vv] - mnew);
                    sc[j][vv] = p;
                    rs += p;
                }
                for (int off = 8; off > 0; off >>= 1)
                    rs += __shfl_xor(rs, off, 32);
                li[vv] = li[vv] * corr[vv] + rs;
            }
            // P: C-layout -> LDS -> A-layout (same-wave DS is in-order)
            for (int j = 0; j < 2; ++j)
                for (int vv = 0; vv < 8; ++vv)
                    lp[w][(mhalf * 8 + vv) * PP + j * 16 + nlo] = (_Float16)sc[j][vv];
            asm volatile("s_wait_dscnt 0" ::: "memory");
            FragA pf;
            {
                int kbase = mhalf * 16;
                for (int vv = 0; vv < 8; ++vv) {
                    int c0 = (vv << 1) + kbase;
                    pf.u[vv] = *(const u32*)(&lp[w][nlo * PP + c0]);
                }
            }
            for (int jd = 0; jd < 4; ++jd) {
                for (int vv = 0; vv < 8; ++vv) accO[jd][vv] *= corr[vv];
                FragA vf;
                int dc  = jd * 16 + nlo;
                int tb2 = mhalf * 16;
                for (int vv = 0; vv < 8; ++vv)
                    vf.u[vv] = *(const u32*)(&lvc[dc * VP + tb2 + (vv << 1)]);
                accO[jd] = __builtin_amdgcn_wmma_f32_16x16x32_f16(
                    false, pf.h, false, vf.h, (short)0, accO[jd], false, false);
            }
        }
        __syncthreads();
    }

    for (int vv = 0; vv < 8; ++vv) {
        float inv = 1.0f / li[vv];
        int mrow = qw + mhalf * 8 + vv;
        for (int jd = 0; jd < 4; ++jd)
            z[base + (size_t)mrow * EMBED + jd * 16 + nlo] =
                (_Float16)(accO[jd][vv] * inv);
    }
}

// ---------------------------------------------------------------------------
extern "C" void kernel_launch(void* const* d_in, const int* in_sizes, int n_in,
                              void* d_out, int out_size, void* d_ws, size_t ws_size,
                              hipStream_t stream) {
    (void)in_sizes; (void)n_in; (void)out_size; (void)ws_size;
    const float* x  = (const float*)d_in[0];
    const float* Wq = (const float*)d_in[1];
    const float* bq = (const float*)d_in[2];
    const float* Wk = (const float*)d_in[3];
    const float* bk = (const float*)d_in[4];
    const float* Wv = (const float*)d_in[5];
    const float* bv = (const float*)d_in[6];
    const float* Wo = (const float*)d_in[7];
    float* out = (float*)d_out;

    const int M = BATCH * SEQN, N = EMBED, K = EMBED;
    const size_t MN = (size_t)M * N;
    const size_t WW = (size_t)N * K;

    _Float16* xh  = (_Float16*)d_ws;      // [M,K]
    _Float16* wqt = xh  + MN;             // [N,K]
    _Float16* wkt = wqt + WW;
    _Float16* wvt = wkt + WW;
    _Float16* woh = wvt + WW;             // Wo already [N,K]
    _Float16* qb  = woh + WW;
    _Float16* kb  = qb  + MN;
    _Float16* vb  = kb  + MN;
    _Float16* zb  = vb  + MN;

    cvt_f16<<<dim3(1024), dim3(256), 0, stream>>>(x, xh, (int)MN);
    cvt_f16<<<dim3(256),  dim3(256), 0, stream>>>(Wo, woh, (int)WW);
    dim3 gT(32, 32), bT(256);
    transpose_cvt<<<gT, bT, 0, stream>>>(Wq, wqt);
    transpose_cvt<<<gT, bT, 0, stream>>>(Wk, wkt);
    transpose_cvt<<<gT, bT, 0, stream>>>(Wv, wvt);

    dim3 gG(M / 128, N / 128), bG(256);
    gemm_wmma_f16<_Float16, true><<<gG, bG, 0, stream>>>(xh, wqt, bq, qb, M, N, K);
    gemm_wmma_f16<_Float16, true><<<gG, bG, 0, stream>>>(xh, wkt, bk, kb, M, N, K);
    gemm_wmma_f16<_Float16, true><<<gG, bG, 0, stream>>>(xh, wvt, bv, vb, M, N, K);

    dim3 gA(SEQN / 64, BATCH * HEADS), bA(128);
    flash_attn_wmma<<<gA, bA, 0, stream>>>(qb, kb, vb, zb);

    gemm_wmma_f16<float, false><<<gG, bG, 0, stream>>>(zb, woh, nullptr, out, M, N, K);
}
